// SpatialAttnBlock_67233418052324
// MI455X (gfx1250) — matrix operand order repaired and checked by reading, not compiled, not run on hardware
//
#include <hip/hip_runtime.h>

// ---------------------------------------------------------------------------
// SpatialAttnBlock on MI455X (gfx1250, wave32, WMMA + Tensor Data Mover).
//  KW: one-shot weight fp32 -> f16 conversion (kills per-tile cvt chains)
//  K0: GroupNorm + SpatialNorm modulation  -> xn  f16 (B,C,HW)
//  K1: Q/K/V GEMMs, 16x64 strip per wave   -> qT (B,HW,C), k (B,C,HW), vT (B,HW,C)
//  K2: flash attention; K tiles staged in LDS by TDM (tensor_load_to_lds,
//      double buffered, s_wait_tensorcnt), WMMA for S=Q^T K and O=P V^T
//  K3: out-projection GEMM + residual      -> d_out f32 (B,C,H,W)
// ---------------------------------------------------------------------------

typedef _Float16 half_t;
typedef __attribute__((ext_vector_type(16))) _Float16 v16h;
typedef __attribute__((ext_vector_type(8)))  _Float16 v8h;
typedef __attribute__((ext_vector_type(8)))  float    v8f;
typedef __attribute__((ext_vector_type(4)))  unsigned int v4u;
typedef __attribute__((ext_vector_type(8)))  int      v8i;
typedef __attribute__((ext_vector_type(4)))  int      v4i;

#define C_    256
#define HW_   4096
#define B_    4
// q scale: C^-0.5 * log2(e)  (softmax done in exp2 domain)
#define QSCALE 0.09016844f

#if __has_builtin(__builtin_amdgcn_tensor_load_to_lds) && \
    __has_builtin(__builtin_amdgcn_s_wait_tensorcnt)
#define HAVE_TDM 1
#else
#define HAVE_TDM 0
#endif

static __device__ __forceinline__ v8f wmma_f16(v16h a, v16h b, v8f c) {
    return __builtin_amdgcn_wmma_f32_16x16x32_f16(false, a, false, b, (short)0, c,
                                                  false, false);
}

// ---------------------------------------------------------------------------
// KW: convert the four 256x256 fp32 weight matrices to f16 once.
// ---------------------------------------------------------------------------
__global__ __launch_bounds__(256)
void kw_convert(const float* __restrict__ wq, const float* __restrict__ wk,
                const float* __restrict__ wv, const float* __restrict__ wo,
                half_t* __restrict__ wh) {
    const int i = blockIdx.x * 256 + threadIdx.x;   // 0 .. 4*65536-1
    const int m = i >> 16, off = i & 65535;
    const float* src = (m == 0) ? wq : (m == 1) ? wk : (m == 2) ? wv : wo;
    wh[i] = (half_t)src[off];
}

// ---------------------------------------------------------------------------
// K0: GroupNorm (32 groups of 8 ch) + zq nearest-upsample modulation -> f16
// ---------------------------------------------------------------------------
__global__ __launch_bounds__(256)
void k0_gn_mod(const float* __restrict__ x, const float* __restrict__ zq,
               const float* __restrict__ gamma, const float* __restrict__ beta,
               const float* __restrict__ wy, const float* __restrict__ by,
               const float* __restrict__ wb, const float* __restrict__ bb,
               half_t* __restrict__ xn) {
    const int bg = blockIdx.x;
    const int b = bg >> 5, g = bg & 31;
    const float* xb = x + (size_t)(b * C_ + g * 8) * HW_;

    float s1 = 0.f, s2 = 0.f;
    for (int idx = threadIdx.x; idx < 8 * HW_; idx += 256) {
        float v = xb[idx];
        s1 += v; s2 += v * v;
    }
#pragma unroll
    for (int m = 16; m >= 1; m >>= 1) {
        s1 += __shfl_xor(s1, m, 32);
        s2 += __shfl_xor(s2, m, 32);
    }
    __shared__ float r1[8], r2[8];
    __shared__ float smu, srstd;
    const int wave = threadIdx.x >> 5, lane = threadIdx.x & 31;
    if (lane == 0) { r1[wave] = s1; r2[wave] = s2; }
    __syncthreads();
    if (threadIdx.x == 0) {
        float a = 0.f, c2 = 0.f;
#pragma unroll
        for (int i = 0; i < 8; ++i) { a += r1[i]; c2 += r2[i]; }
        const float inv_n = 1.0f / (8.0f * HW_);
        float mu = a * inv_n;
        float var = c2 * inv_n - mu * mu;
        smu = mu;
        srstd = rsqrtf(var + 1e-6f);
    }
    __syncthreads();
    const float mu = smu, rs = srstd;

    for (int idx = threadIdx.x; idx < 8 * HW_; idx += 256) {
        const int cl = idx >> 12;
        const int p  = idx & (HW_ - 1);
        const int c  = g * 8 + cl;
        const int ph = p >> 6, pw = p & 63;
        const int zb = b * 1024 + (ph >> 2) * 16 + (pw >> 2);
        const float z0 = zq[zb], z1 = zq[zb + 256], z2 = zq[zb + 512], z3 = zq[zb + 768];
        const float my = by[c] + wy[c * 4] * z0 + wy[c * 4 + 1] * z1 +
                         wy[c * 4 + 2] * z2 + wy[c * 4 + 3] * z3;
        const float mb = bb[c] + wb[c * 4] * z0 + wb[c * 4 + 1] * z1 +
                         wb[c * 4 + 2] * z2 + wb[c * 4 + 3] * z3;
        const float v = xb[idx];
        const float xv = ((v - mu) * rs * gamma[c] + beta[c]) * my + mb;
        xn[(size_t)(b * C_ + c) * HW_ + p] = (half_t)xv;
    }
}

// ---------------------------------------------------------------------------
// K1: Q/K/V projections. One wave per 16(oc) x 64(pixels) strip: A-frag loaded
// once per K-step, reused across 4 B-frags (4 accumulators, 32 wmma / wave).
// waves = 3 * B * (C/16) * (HW/64) = 12288; 4 waves / block.
// ---------------------------------------------------------------------------
__global__ __launch_bounds__(128)
void k1_qkv(const half_t* __restrict__ wh,      // f16 weights: q,k,v,(o)
            const float* __restrict__ bq, const float* __restrict__ bk,
            const float* __restrict__ bv,
            const half_t* __restrict__ xn,
            half_t* __restrict__ qT, half_t* __restrict__ kM,
            half_t* __restrict__ vT) {
    int wg = blockIdx.x * 4 + (threadIdx.x >> 5);
    const int lane = threadIdx.x & 31;
    const int jt = wg & 63;   wg >>= 6;
    const int oct = wg & 15;  wg >>= 4;
    const int bidx = wg & 3;  wg >>= 2;
    const int m = wg;  // 0=q 1=k 2=v
    const half_t* W   = wh + (size_t)m * C_ * C_;
    const float* bias = (m == 0) ? bq : (m == 1) ? bk : bv;
    const int oc0 = oct * 16, j0 = jt * 64;
    const half_t* xb = xn + (size_t)bidx * C_ * HW_;
    const int r = lane & 15, hf = lane >> 4;

    v8f acc[4] = {};
#pragma unroll
    for (int kk = 0; kk < C_; kk += 32) {
        const half_t* wrow = W + (size_t)(oc0 + r) * C_ + kk + hf * 8;
        v8h alo = *(const v8h*)wrow;
        v8h ahi = *(const v8h*)(wrow + 16);
        v16h a;
#pragma unroll
        for (int i = 0; i < 8; ++i) { a[i] = alo[i]; a[i + 8] = ahi[i]; }
        const half_t* brow = xb + (size_t)(kk + hf * 8) * HW_ + j0 + r;
#pragma unroll
        for (int s = 0; s < 4; ++s) {
            v16h bfrag;
#pragma unroll
            for (int e = 0; e < 16; ++e)
                bfrag[e] = brow[(size_t)(e < 8 ? e : e + 8) * HW_ + s * 16];
            acc[s] = wmma_f16(a, bfrag, acc[s]);
        }
    }

    const int n = r, mr = hf * 8;
    if (m == 0) {                       // q -> qT (HW, C), scaled, contiguous
#pragma unroll
        for (int s = 0; s < 4; ++s) {
            v8h out;
#pragma unroll
            for (int i = 0; i < 8; ++i)
                out[i] = (_Float16)((acc[s][i] + bias[oc0 + mr + i]) * QSCALE);
            *(v8h*)(qT + (size_t)bidx * HW_ * C_ +
                    (size_t)(j0 + s * 16 + n) * C_ + oc0 + mr) = out;
        }
    } else if (m == 1) {                // k -> (C, HW)
#pragma unroll
        for (int s = 0; s < 4; ++s)
#pragma unroll
            for (int i = 0; i < 8; ++i)
                kM[(size_t)bidx * C_ * HW_ + (size_t)(oc0 + mr + i) * HW_ +
                   j0 + s * 16 + n] = (_Float16)(acc[s][i] + bias[oc0 + mr + i]);
    } else {                            // v -> vT (HW, C), contiguous
#pragma unroll
        for (int s = 0; s < 4; ++s) {
            v8h out;
#pragma unroll
            for (int i = 0; i < 8; ++i)
                out[i] = (_Float16)(acc[s][i] + bias[oc0 + mr + i]);
            *(v8h*)(vT + (size_t)bidx * HW_ * C_ +
                    (size_t)(j0 + s * 16 + n) * C_ + oc0 + mr) = out;
        }
    }
}

// ---------------------------------------------------------------------------
// TDM: DMA one 256(C) x 32(j) f16 K tile (row-major, 80 B padded LDS rows)
// from global (C,HW) layout into LDS.  D# built per CDNA5 ISA §8.3-8.6.
// This toolchain exposes the 6-arg builtin (extra int32x8 group, then cpol).
// ---------------------------------------------------------------------------
#if HAVE_TDM
static __device__ __forceinline__ void tdm_load_k_tile(const half_t* gsrc,
                                                       unsigned lds_off) {
    const unsigned long long ga = (unsigned long long)(uintptr_t)gsrc;
    v4u g0;
    g0.x = 1u;                                            // count=1 (valid user D#)
    g0.y = lds_off;                                       // LDS byte address
    g0.z = (unsigned)(ga & 0xFFFFFFFFu);                  // global_addr[31:0]
    g0.w = (unsigned)((ga >> 32) & 0x01FFFFFFu) | (2u << 30);  // addr[56:32], type=2
    v8i g1;
    // data_size=2B(1<<16) | pad_enable(1<<20) | pad_interval=3 (64B) <<22
    // | pad_amount=3 (16B) <<25   -> 80B LDS row stride (bank-friendly)
    g1[0] = 0x06D10000;
    g1[1] = (int)(4096u << 16);      // tensor_dim0 = 4096 (bits 79:48, low half)
    g1[2] = (int)(256u  << 16);      // tensor_dim0 hi | tensor_dim1 = 256
    g1[3] = (int)(32u   << 16);      // tensor_dim1 hi | tile_dim0 = 32
    g1[4] = 256;                     // tile_dim1 = 256, tile_dim2 = 0
    g1[5] = 4096;                    // tensor_dim0_stride = 4096 elements
    g1[6] = 0;                       // stride hi | tensor_dim1_stride lo
    g1[7] = 0;
    v4i gz4 = {0, 0, 0, 0};          // groups 2/3 unused (2D tensor)
    v8i gz8 = {0, 0, 0, 0, 0, 0, 0, 0};
    __builtin_amdgcn_tensor_load_to_lds(g0, g1, gz4, gz4, gz8, 0);
}
#endif

// ---------------------------------------------------------------------------
// K2: flash attention. 1 block = 16 queries, 8 waves; wave w owns output
// channels [32w, 32w+32).  K tiles are DMA'd once per block into LDS by the
// Tensor Data Mover (double buffered, overlapped with compute); every wave
// computes the 16x32 S tile from LDS, runs the online softmax with 16-lane
// shuffle reductions, relays P through its private LDS slice, then does
// O += P * V^T with WMMA on its channel slice.
// ---------------------------------------------------------------------------
__global__ __launch_bounds__(256)
void k2_attn(const half_t* __restrict__ qT, const half_t* __restrict__ kM,
             const half_t* __restrict__ vT, half_t* __restrict__ aT) {
#if HAVE_TDM
    __shared__ _Float16 Klds[2][256][40];   // 2 x 20 KB, 80 B rows (TDM padded)
#endif
    __shared__ _Float16 Plds[8][16][32];    // per-wave private slices, 8 KB

    const int bidx = blockIdx.x >> 8;
    const int q0 = (blockIdx.x & 255) * 16;
    const int wave = threadIdx.x >> 5;
    const int lane = threadIdx.x & 31;
    const int r = lane & 15, hf = lane >> 4;
    const int cbase = wave * 32;

    const half_t* qb = qT + (size_t)bidx * HW_ * C_;
    const half_t* kb = kM + (size_t)bidx * C_ * HW_;
    const half_t* vb = vT + (size_t)bidx * HW_ * C_;

    // preload Q A-fragments (16 queries x 256 channels)
    v16h qa[8];
#pragma unroll
    for (int kk = 0; kk < 8; ++kk) {
        const half_t* p = qb + (size_t)(q0 + r) * C_ + kk * 32 + hf * 8;
        v8h lo = *(const v8h*)p;
        v8h hi = *(const v8h*)(p + 16);
#pragma unroll
        for (int i = 0; i < 8; ++i) { qa[kk][i] = lo[i]; qa[kk][i + 8] = hi[i]; }
    }

    float mrow[8], lrow[8];
    v8f o0 = {}, o1 = {};
#pragma unroll
    for (int i = 0; i < 8; ++i) { mrow[i] = -1e30f; lrow[i] = 0.f; }

#if HAVE_TDM
    const bool issuer = (wave == 0);
    if (issuer)
        tdm_load_k_tile(kb, (unsigned)(uintptr_t)&Klds[0][0][0]);
#endif

    for (int jb = 0; jb < HW_; jb += 32) {
#if HAVE_TDM
        const int cur = (jb >> 5) & 1;
        if (issuer) {
            if (jb + 32 < HW_) {
                tdm_load_k_tile(kb + jb + 32,
                                (unsigned)(uintptr_t)&Klds[cur ^ 1][0][0]);
                __builtin_amdgcn_s_wait_tensorcnt(1);   // current tile landed
            } else {
                __builtin_amdgcn_s_wait_tensorcnt(0);
            }
        }
        __syncthreads();                                // tile visible to all waves
#else
        if (jb + 32 < HW_)
            __builtin_prefetch(kb + (size_t)(r + hf * 16) * HW_ + jb + 32, 0, 3);
#endif
        if (jb + 32 < HW_)
            __builtin_prefetch(vb + (size_t)(jb + 32 + r + hf * 16) * C_ + cbase, 0, 3);

        // ---- S = Q^T K tile (16 x 32), f32 accum -------------------------
        v8f s0 = {}, s1 = {};
#pragma unroll
        for (int kk = 0; kk < 8; ++kk) {
#if HAVE_TDM
            const _Float16* bp = &Klds[cur][kk * 32 + hf * 8][r];
            v16h b0, b1;
#pragma unroll
            for (int i = 0; i < 8; ++i) {
                b0[i] = bp[i * 40]; b0[i + 8] = bp[(i + 16) * 40];
                b1[i] = bp[i * 40 + 16]; b1[i + 8] = bp[(i + 16) * 40 + 16];
            }
#else
            const half_t* gp = kb + (size_t)(kk * 32 + hf * 8) * HW_ + jb + r;
            v16h b0, b1;
#pragma unroll
            for (int e = 0; e < 16; ++e) {
                b0[e] = gp[(size_t)(e < 8 ? e : e + 8) * HW_];
                b1[e] = gp[(size_t)(e < 8 ? e : e + 8) * HW_ + 16];
            }
#endif
            s0 = wmma_f16(qa[kk], b0, s0);
            s1 = wmma_f16(qa[kk], b1, s1);
        }
#if HAVE_TDM
        __syncthreads();     // all waves done reading Klds[cur] before reuse
#endif
        // ---- online softmax (exp2 domain; scale folded into Q) -----------
        float cf[8];
#pragma unroll
        for (int i = 0; i < 8; ++i) {
            float t = fmaxf(s0[i], s1[i]);
#pragma unroll
            for (int m = 8; m >= 1; m >>= 1) t = fmaxf(t, __shfl_xor(t, m, 32));
            const float mn = fmaxf(mrow[i], t);
            cf[i] = exp2f(mrow[i] - mn);
            mrow[i] = mn;
        }
        v8f p0, p1;
#pragma unroll
        for (int i = 0; i < 8; ++i) {
            p0[i] = exp2f(s0[i] - mrow[i]);
            p1[i] = exp2f(s1[i] - mrow[i]);
            float rs = p0[i] + p1[i];
#pragma unroll
            for (int m = 8; m >= 1; m >>= 1) rs += __shfl_xor(rs, m, 32);
            lrow[i] = lrow[i] * cf[i] + rs;
            o0[i] *= cf[i];
            o1[i] *= cf[i];
        }
        // ---- P: D-layout -> A-layout via this wave's LDS slice -----------
        const int mr = hf * 8;
#pragma unroll
        for (int i = 0; i < 8; ++i) {
            Plds[wave][mr + i][r]      = (_Float16)p0[i];
            Plds[wave][mr + i][r + 16] = (_Float16)p1[i];
        }
        __builtin_amdgcn_wave_barrier();
        v16h pa;
#pragma unroll
        for (int i = 0; i < 8; ++i) {
            pa[i]     = Plds[wave][r][hf * 8 + i];
            pa[i + 8] = Plds[wave][r][16 + hf * 8 + i];
        }
        // ---- O += P * V^T for this wave's 32 channels --------------------
        const half_t* vp = vb + (size_t)(jb + hf * 8) * C_ + cbase + r;
        v16h vb0, vb1;
#pragma unroll
        for (int e = 0; e < 16; ++e) {
            vb0[e] = vp[(size_t)(e < 8 ? e : e + 8) * C_];
            vb1[e] = vp[(size_t)(e < 8 ? e : e + 8) * C_ + 16];
        }
        o0 = wmma_f16(pa, vb0, o0);
        o1 = wmma_f16(pa, vb1, o1);
    }

    // ---- normalize and store attn-out as (C, HW) f16 ---------------------
    v8h w0, w1;
#pragma unroll
    for (int i = 0; i < 8; ++i) {
        const float inv = 1.0f / lrow[i];
        w0[i] = (_Float16)(o0[i] * inv);
        w1[i] = (_Float16)(o1[i] * inv);
    }
    half_t* ab = aT + (size_t)bidx * C_ * HW_;
    *(v8h*)(ab + (size_t)(cbase + r) * HW_ + q0 + hf * 8)      = w0;
    *(v8h*)(ab + (size_t)(cbase + 16 + r) * HW_ + q0 + hf * 8) = w1;
}

// ---------------------------------------------------------------------------
// K3: out = x + (wo @ attn_out + bo). One wave per 16(oc) x 64(pixels) strip.
// ---------------------------------------------------------------------------
__global__ __launch_bounds__(128)
void k3_proj(const half_t* __restrict__ who, const float* __restrict__ bo,
             const half_t* __restrict__ aT, const float* __restrict__ x,
             float* __restrict__ out) {
    int wg = blockIdx.x * 4 + (threadIdx.x >> 5);
    const int lane = threadIdx.x & 31;
    const int jt = wg & 63;   wg >>= 6;
    const int oct = wg & 15;  wg >>= 4;
    const int bidx = wg & 3;
    const int oc0 = oct * 16, j0 = jt * 64;
    const int r = lane & 15, hf = lane >> 4;
    const half_t* ab = aT + (size_t)bidx * C_ * HW_;

    v8f acc[4] = {};
#pragma unroll
    for (int kk = 0; kk < C_; kk += 32) {
        const half_t* wrow = who + (size_t)(oc0 + r) * C_ + kk + hf * 8;
        v8h alo = *(const v8h*)wrow;
        v8h ahi = *(const v8h*)(wrow + 16);
        v16h a;
#pragma unroll
        for (int i = 0; i < 8; ++i) { a[i] = alo[i]; a[i + 8] = ahi[i]; }
        const half_t* brow = ab + (size_t)(kk + hf * 8) * HW_ + j0 + r;
#pragma unroll
        for (int s = 0; s < 4; ++s) {
            v16h bfrag;
#pragma unroll
            for (int e = 0; e < 16; ++e)
                bfrag[e] = brow[(size_t)(e < 8 ? e : e + 8) * HW_ + s * 16];
            acc[s] = wmma_f16(a, bfrag, acc[s]);
        }
    }
    const int n = r, mr = hf * 8;
#pragma unroll
    for (int s = 0; s < 4; ++s)
#pragma unroll
        for (int i = 0; i < 8; ++i) {
            const int c = oc0 + mr + i;
            const size_t off = (size_t)(bidx * C_ + c) * HW_ + j0 + s * 16 + n;
            out[off] = x[off] + acc[s][i] + bo[c];
        }
}

// ---------------------------------------------------------------------------
extern "C" void kernel_launch(void* const* d_in, const int* in_sizes, int n_in,
                              void* d_out, int out_size, void* d_ws, size_t ws_size,
                              hipStream_t stream) {
    (void)in_sizes; (void)n_in; (void)out_size; (void)ws_size;
    const float* x      = (const float*)d_in[0];
    const float* zq     = (const float*)d_in[1];
    const float* gamma  = (const float*)d_in[2];
    const float* beta   = (const float*)d_in[3];
    const float* wy     = (const float*)d_in[4];
    const float* by     = (const float*)d_in[5];
    const float* wb     = (const float*)d_in[6];
    const float* bb     = (const float*)d_in[7];
    const float* wq     = (const float*)d_in[8];
    const float* bq     = (const float*)d_in[9];
    const float* wk     = (const float*)d_in[10];
    const float* bk     = (const float*)d_in[11];
    const float* wv     = (const float*)d_in[12];
    const float* bv     = (const float*)d_in[13];
    const float* wo     = (const float*)d_in[14];
    const float* bo     = (const float*)d_in[15];
    float* out = (float*)d_out;

    // workspace: 5 f16 tensors of B*C*HW (8 MiB each) + 4 f16 weight matrices
    const size_t NE = (size_t)B_ * C_ * HW_;
    half_t* xn = (half_t*)d_ws;
    half_t* qT = xn + NE;
    half_t* kM = qT + NE;
    half_t* vT = kM + NE;
    half_t* aT = vT + NE;
    half_t* wh = aT + NE;                 // wq|wk|wv|wo, 65536 halves each
    half_t* who = wh + 3 * 65536;

    kw_convert<<<1024, 256, 0, stream>>>(wq, wk, wv, wo, wh);
    k0_gn_mod<<<B_ * 32, 256, 0, stream>>>(x, zq, gamma, beta, wy, by, wb, bb, xn);
    k1_qkv<<<(3 * B_ * 16 * 64) / 4, 128, 0, stream>>>(wh, bq, bk, bv, xn,
                                                       qT, kM, vT);
    k2_attn<<<B_ * 256, 256, 0, stream>>>(qT, kM, vT, aT);
    k3_proj<<<(B_ * 16 * 64) / 4, 128, 0, stream>>>(who, bo, aT, x, out);
}
